// CrossAttention_78176994722426
// MI455X (gfx1250) — compile-verified, matrix operand order checked by reading, not run on hardware
//
#include <hip/hip_runtime.h>
#include <hip/hip_bf16.h>

// ---------------------------------------------------------------------------
// Cross-attention for MI455X (gfx1250, wave32, WMMA, async-to-LDS).
//
// ~86 GFLOP vs ~50MB data -> compute bound (23.3 TB/s HBM; fits in 192MB L2).
// All matmuls via v_wmma_f32_16x16x32_f16. fp32->f16 conversion done once by
// an elementwise kernel so GEMM inner loops move pure f16 bytes, staged into
// double-buffered LDS with GLOBAL_LOAD_ASYNC_TO_LDS_B128 (ASYNCcnt pipeline).
// ---------------------------------------------------------------------------

typedef float     v8f  __attribute__((ext_vector_type(8)));
typedef _Float16  v8h  __attribute__((ext_vector_type(8)));
typedef _Float16  v16h __attribute__((ext_vector_type(16)));

__device__ __forceinline__ v8f wmma_f32_f16(v16h a, v16h b, v8f c) {
  return __builtin_amdgcn_wmma_f32_16x16x32_f16(false, a, false, b, (short)0, c,
                                                false, false);
}

__device__ __forceinline__ v8f zero_v8f() {
  v8f z;
#pragma unroll
  for (int i = 0; i < 8; ++i) z[i] = 0.0f;
  return z;
}

// A-fragment (16x32 f16) / transposed-A-style B-fragment load from LDS.
// ISA 7.12.2: lane l -> row l%16, g=l/16; half h -> k = h + 8g (+8 if h>=8).
// Two contiguous 16B chunks -> ds_load_b128 (strides are 16B-divisible).
__device__ __forceinline__ v16h load_frag(const _Float16* base, int stride,
                                          int row, int koff, int g) {
  const _Float16* p = base + row * stride + koff + 8 * g;
  v8h lo = *(const v8h*)(p);
  v8h hi = *(const v8h*)(p + 16);
  v16h r;
#pragma unroll
  for (int e = 0; e < 8; ++e) { r[e] = lo[e]; r[e + 8] = hi[e]; }
  return r;
}

// Async global->LDS 16-byte copy (per lane). Tracked by ASYNCcnt.
// saddr form: uniform 64-bit base in SGPR pair, 32-bit per-lane byte offset,
// LDS byte offset in vdst VGPR (generic LDS addr low 32 bits == LDS offset).
__device__ __forceinline__ void async_copy_b128(unsigned lds_off,
                                                const void* sbase,
                                                unsigned voff) {
  asm volatile("global_load_async_to_lds_b128 %0, %1, %2"
               :
               : "v"(lds_off), "v"(voff), "s"(sbase)
               : "memory");
}

__device__ __forceinline__ void wait_async0() {
  asm volatile("s_wait_asynccnt 0x0" ::: "memory");
}

// ---------------------------------------------------------------------------
// Elementwise fp32 -> f16 (one-time; makes every GEMM/attention input f16).
// ---------------------------------------------------------------------------
__global__ __launch_bounds__(256) void cvt_f32_f16_kernel(
    const float* __restrict__ src, _Float16* __restrict__ dst, int n8) {
  const int i = blockIdx.x * blockDim.x + threadIdx.x;
  if (i < n8) {
    v8f x = *(const v8f*)(src + (size_t)i * 8);
    v8h h;
#pragma unroll
    for (int e = 0; e < 8; ++e) h[e] = (_Float16)x[e];
    *(v8h*)(dst + (size_t)i * 8) = h;
  }
}

// ---------------------------------------------------------------------------
// GEMM: Y[m,n] = sum_k X[m,k]*W[n,k] + bias[n]  (X,W f16; Y f16 or f32).
// Block 256 = 8 waves; tile 128(M) x 128(N); wave tile 64x32 (c[4][2]).
// K stepped by 32 with DOUBLE-BUFFERED async-to-LDS staging:
//   issue async(buf^1, kt+32) -> wmma on buf -> s_wait_asynccnt 0 -> barrier.
// 8 wmma per wave per k-step from 6 ds_load_b128 fragment loads.
// ---------------------------------------------------------------------------
template <typename OutT>
__global__ __launch_bounds__(256) void gemm_h_kernel(
    const _Float16* __restrict__ X, const _Float16* __restrict__ W,
    const float* __restrict__ bias, OutT* __restrict__ Y,
    int M, int N, int K) {
  const int n0 = blockIdx.x * 128;
  const int m0 = blockIdx.y * 128;
  const int t  = threadIdx.x;
  const int w  = t >> 5;
  const int l  = t & 31;
  const int lm = l & 15;
  const int g  = l >> 4;
  const int wm = w & 1;    // 2 waves along M (64 rows each)
  const int wn = w >> 1;   // 4 waves along N (32 cols each)

  // stride 40 halfs = 80B (16B-divisible -> b128 frag loads); 2 buffers.
  __shared__ _Float16 As[2][128 * 40];
  __shared__ _Float16 Bs[2][128 * 40];

  v8f c[4][2];
#pragma unroll
  for (int i = 0; i < 4; ++i)
#pragma unroll
    for (int j = 0; j < 2; ++j) c[i][j] = zero_v8f();

  // Per-thread staging share: one row (t>>1) of the 128x32 slab, 16-half
  // column chunk ((t&1)*16) -> two async b128 per matrix per k-step.
  const int srow = t >> 1;
  const int scol = (t & 1) * 16;
  const unsigned grow_off = (unsigned)((srow * K + scol) * 2);  // bytes
  const unsigned lrow_off = (unsigned)((srow * 40 + scol) * 2); // bytes

  const _Float16* Xb = X + (size_t)m0 * K;
  const _Float16* Wb = W + (size_t)n0 * K;

  // Prologue: stage slab 0 into buffer 0.
  {
    unsigned la = (unsigned)(uintptr_t)(&As[0][0]) + lrow_off;
    unsigned lb = (unsigned)(uintptr_t)(&Bs[0][0]) + lrow_off;
    async_copy_b128(la,       Xb, grow_off);
    async_copy_b128(la + 16u, Xb, grow_off + 16u);
    async_copy_b128(lb,       Wb, grow_off);
    async_copy_b128(lb + 16u, Wb, grow_off + 16u);
  }
  wait_async0();
  __syncthreads();

  for (int kt = 0; kt < K; kt += 32) {
    const int cur = (kt >> 5) & 1;
    const bool more = (kt + 32) < K;

    // Issue next slab into the other buffer (overlaps with wmma below).
    if (more) {
      const unsigned gofs = grow_off + (unsigned)((kt + 32) * 2);
      unsigned la = (unsigned)(uintptr_t)(&As[cur ^ 1][0]) + lrow_off;
      unsigned lb = (unsigned)(uintptr_t)(&Bs[cur ^ 1][0]) + lrow_off;
      async_copy_b128(la,       Xb, gofs);
      async_copy_b128(la + 16u, Xb, gofs + 16u);
      async_copy_b128(lb,       Wb, gofs);
      async_copy_b128(lb + 16u, Wb, gofs + 16u);
    }

    // Compute on current buffer: 4 A-frags, 2 B-frags, 8 wmma.
    const _Float16* Ab = As[cur];
    const _Float16* Bb = Bs[cur];
    v16h af[4];
#pragma unroll
    for (int i = 0; i < 4; ++i)
      af[i] = load_frag(Ab, 40, wm * 64 + i * 16 + lm, 0, g);
#pragma unroll
    for (int j = 0; j < 2; ++j) {
      v16h bf = load_frag(Bb, 40, wn * 32 + j * 16 + lm, 0, g);
#pragma unroll
      for (int i = 0; i < 4; ++i) c[i][j] = wmma_f32_f16(af[i], bf, c[i][j]);
    }

    if (more) wait_async0();
    __syncthreads();
  }

  // Epilogue: C layout -> lane holds col n = lm, rows r + 8g.
#pragma unroll
  for (int j = 0; j < 2; ++j) {
    const int ncol = n0 + wn * 32 + j * 16 + lm;
    const float bj = bias[ncol];
#pragma unroll
    for (int i = 0; i < 4; ++i) {
#pragma unroll
      for (int r = 0; r < 8; ++r) {
        const int mrow = m0 + wm * 64 + i * 16 + r + 8 * g;
        Y[(size_t)mrow * N + ncol] = (OutT)(c[i][j][r] + bj);
      }
    }
  }
}

// ---------------------------------------------------------------------------
// Flash attention (online softmax), f16 q/k/v, f32 accumulate, f16 ctx out.
// Grid: (LQ/64, H, B); block 128 = 4 waves; wave owns 16 query rows.
// KV chunked by 32: S(16x32)=4 wmma, PV(16x64)=4 wmma per wave per chunk.
// ---------------------------------------------------------------------------
#define AB   4
#define ALQ  1024
#define ALKV 2048
#define AD   1024
#define AH   16
#define AHD  64

__global__ __launch_bounds__(128) void attn_kernel(
    const _Float16* __restrict__ q, const _Float16* __restrict__ kmat,
    const _Float16* __restrict__ vmat, _Float16* __restrict__ ctx,
    const float* __restrict__ scale_p, const float* __restrict__ beta_p) {
  const int qt = blockIdx.x;
  const int h  = blockIdx.y;
  const int b  = blockIdx.z;
  const int t  = threadIdx.x;
  const int w  = t >> 5;
  const int l  = t & 31;
  const int lm = l & 15;
  const int g  = l >> 4;

  __shared__ _Float16 Qs[64 * 72];     // 64 q-rows x 64 d (stride 72)
  __shared__ _Float16 Ks[32 * 72];     // 32 kv-rows x 64 d
  __shared__ _Float16 Vt[64 * 40];     // transposed: 64 d x 32 kv
  __shared__ _Float16 Ps[4][16 * 32];  // per-wave P scratch (C->A relayout)

  const float scale = scale_p[0] * 0.125f;  // 1/sqrt(64)
  const float beta  = beta_p[0];

  {  // Stage Q tile once.
    const int row = t >> 1;
    const int cb  = (t & 1) * 32;
    const _Float16* src =
        q + ((size_t)(b * ALQ + qt * 64 + row)) * AD + h * AHD + cb;
    _Float16* dst = &Qs[row * 72 + cb];
#pragma unroll
    for (int i = 0; i < 4; ++i)
      *(v8h*)(dst + i * 8) = *(const v8h*)(src + i * 8);
  }
  __syncthreads();

  v16h qa0 = load_frag(Qs, 72, w * 16 + lm, 0, g);
  v16h qa1 = load_frag(Qs, 72, w * 16 + lm, 32, g);

  float mrow[8], lrow[8];
  v8f acc[4];
#pragma unroll
  for (int r = 0; r < 8; ++r) { mrow[r] = -1e30f; lrow[r] = 0.0f; }
#pragma unroll
  for (int d = 0; d < 4; ++d) acc[d] = zero_v8f();

  for (int cchunk = 0; cchunk < ALKV / 32; ++cchunk) {
    const int kv0 = cchunk * 32;
    __syncthreads();

    {  // K chunk row-major.
      const int row = t >> 2;
      const int cb  = (t & 3) * 16;
      const _Float16* src =
          kmat + ((size_t)(b * ALKV + kv0 + row)) * AD + h * AHD + cb;
      *(v8h*)(&Ks[row * 72 + cb])     = *(const v8h*)(src);
      *(v8h*)(&Ks[row * 72 + cb + 8]) = *(const v8h*)(src + 8);
    }
    {  // V chunk transposed.
      const int row = t >> 2;
      const int cb  = (t & 3) * 16;
      const _Float16* src =
          vmat + ((size_t)(b * ALKV + kv0 + row)) * AD + h * AHD + cb;
#pragma unroll
      for (int i = 0; i < 16; ++i) Vt[(cb + i) * 40 + row] = src[i];
    }
    __syncthreads();

    v8f s0 = zero_v8f(), s1 = zero_v8f();
    {
      v16h kb;
      kb = load_frag(Ks, 72, lm, 0, g);        s0 = wmma_f32_f16(qa0, kb, s0);
      kb = load_frag(Ks, 72, lm, 32, g);       s0 = wmma_f32_f16(qa1, kb, s0);
      kb = load_frag(Ks, 72, 16 + lm, 0, g);   s1 = wmma_f32_f16(qa0, kb, s1);
      kb = load_frag(Ks, 72, 16 + lm, 32, g);  s1 = wmma_f32_f16(qa1, kb, s1);
    }

    float p0[8], p1[8];
#pragma unroll
    for (int r = 0; r < 8; ++r) {
      const float a0 = s0[r] * scale + beta;
      const float a1 = s1[r] * scale + beta;
      float mx = fmaxf(a0, a1);
      mx = fmaxf(mx, __shfl_xor(mx, 1, 32));
      mx = fmaxf(mx, __shfl_xor(mx, 2, 32));
      mx = fmaxf(mx, __shfl_xor(mx, 4, 32));
      mx = fmaxf(mx, __shfl_xor(mx, 8, 32));
      const float mnew  = fmaxf(mrow[r], mx);
      const float alpha = __expf(mrow[r] - mnew);
      mrow[r] = mnew;
      p0[r] = __expf(a0 - mnew);
      p1[r] = __expf(a1 - mnew);
      lrow[r] = lrow[r] * alpha + p0[r] + p1[r];
#pragma unroll
      for (int d = 0; d < 4; ++d) acc[d][r] *= alpha;
    }

#pragma unroll
    for (int r = 0; r < 8; ++r) {
      const int m = r + 8 * g;
      Ps[w][m * 32 + lm]      = (_Float16)p0[r];
      Ps[w][m * 32 + 16 + lm] = (_Float16)p1[r];
    }
    v16h pa = load_frag(Ps[w], 32, lm, 0, g);

#pragma unroll
    for (int d = 0; d < 4; ++d) {
      v16h vb = load_frag(Vt, 40, d * 16 + lm, 0, g);
      acc[d] = wmma_f32_f16(pa, vb, acc[d]);
    }
  }

#pragma unroll
  for (int r = 0; r < 8; ++r) {
    float s = lrow[r];
    s += __shfl_xor(s, 1, 32);
    s += __shfl_xor(s, 2, 32);
    s += __shfl_xor(s, 4, 32);
    s += __shfl_xor(s, 8, 32);
    const float rinv = 1.0f / s;
    const int qrow = qt * 64 + w * 16 + r + 8 * g;
#pragma unroll
    for (int d = 0; d < 4; ++d) {
      const int col = h * AHD + d * 16 + lm;
      ctx[((size_t)(b * ALQ + qrow)) * AD + col] = (_Float16)(acc[d][r] * rinv);
    }
  }
}

// ---------------------------------------------------------------------------
// Launcher. Workspace (f16 unless noted):
//  [0,8)MB qdom | [8,24) kvdom | [24,26) Wq | [26,28) Wk | [28,30) Wv
//  [30,32) Wo | [32,40) q | [40,56) k | [56,72) v | [72,80) ctx  => 80MB
// ---------------------------------------------------------------------------
extern "C" void kernel_launch(void* const* d_in, const int* in_sizes, int n_in,
                              void* d_out, int out_size, void* d_ws,
                              size_t ws_size, hipStream_t stream) {
  (void)in_sizes; (void)n_in; (void)out_size; (void)ws_size;

  const float* qdom  = (const float*)d_in[0];
  const float* kvdom = (const float*)d_in[1];
  const float* Wq = (const float*)d_in[2];
  const float* bq = (const float*)d_in[3];
  const float* Wk = (const float*)d_in[4];
  const float* bk = (const float*)d_in[5];
  const float* Wv = (const float*)d_in[6];
  const float* bv = (const float*)d_in[7];
  const float* Wo = (const float*)d_in[8];
  const float* bo = (const float*)d_in[9];
  const float* scale = (const float*)d_in[10];
  const float* beta  = (const float*)d_in[11];
  float* out = (float*)d_out;

  const size_t MB = 1u << 20;
  char* ws = (char*)d_ws;
  _Float16* qdom_h  = (_Float16*)(ws + 0 * MB);
  _Float16* kvdom_h = (_Float16*)(ws + 8 * MB);
  _Float16* Wq_h    = (_Float16*)(ws + 24 * MB);
  _Float16* Wk_h    = (_Float16*)(ws + 26 * MB);
  _Float16* Wv_h    = (_Float16*)(ws + 28 * MB);
  _Float16* Wo_h    = (_Float16*)(ws + 30 * MB);
  _Float16* qh      = (_Float16*)(ws + 32 * MB);
  _Float16* kh      = (_Float16*)(ws + 40 * MB);
  _Float16* vh      = (_Float16*)(ws + 56 * MB);
  _Float16* ctxh    = (_Float16*)(ws + 72 * MB);

  auto cvt = [&](const float* s, _Float16* d, size_t n) {
    const int n8 = (int)(n / 8);
    cvt_f32_f16_kernel<<<dim3((n8 + 255) / 256), dim3(256), 0, stream>>>(s, d,
                                                                         n8);
  };
  cvt(qdom,  qdom_h,  (size_t)4096 * 1024);
  cvt(kvdom, kvdom_h, (size_t)8192 * 1024);
  cvt(Wq, Wq_h, (size_t)1024 * 1024);
  cvt(Wk, Wk_h, (size_t)1024 * 1024);
  cvt(Wv, Wv_h, (size_t)1024 * 1024);
  cvt(Wo, Wo_h, (size_t)1024 * 1024);

  const dim3 blk(256);
  gemm_h_kernel<_Float16><<<dim3(1024 / 128, 4096 / 128), blk, 0, stream>>>(
      qdom_h, Wq_h, bq, qh, 4096, 1024, 1024);
  gemm_h_kernel<_Float16><<<dim3(1024 / 128, 8192 / 128), blk, 0, stream>>>(
      kvdom_h, Wk_h, bk, kh, 8192, 1024, 1024);
  gemm_h_kernel<_Float16><<<dim3(1024 / 128, 8192 / 128), blk, 0, stream>>>(
      kvdom_h, Wv_h, bv, vh, 8192, 1024, 1024);

  attn_kernel<<<dim3(ALQ / 64, AH, AB), dim3(128), 0, stream>>>(
      qh, kh, vh, ctxh, scale, beta);

  gemm_h_kernel<float><<<dim3(1024 / 128, 4096 / 128), blk, 0, stream>>>(
      ctxh, Wo_h, bo, out, 4096, 1024, 1024);
}
